// SparseAttention_79156247265918
// MI455X (gfx1250) — matrix operand order repaired and checked by reading, not compiled, not run on hardware
//
#include <hip/hip_runtime.h>

typedef __attribute__((ext_vector_type(16))) _Float16 v16h;
typedef __attribute__((ext_vector_type(8)))  float    v8f;
typedef unsigned int v4u  __attribute__((ext_vector_type(4)));
typedef int          v8i_ __attribute__((ext_vector_type(8)));
typedef int          v4i_ __attribute__((ext_vector_type(4)));

#define B_    2
#define H_    12
#define S_    2048
#define D_    64
#define E_    8
#define KB    32            // keys per iteration
#define WAVES 4             // waves per workgroup
#define QT    16            // query rows per wave
#define NTILE (S_ / KB)     // 64 key tiles

#if __has_builtin(__builtin_amdgcn_tensor_load_to_lds)
#define HAVE_TDM 1
#else
#define HAVE_TDM 0
#endif

#if HAVE_TDM
// ---------- generic 2-D TDM descriptor (f16 elements, padded LDS pitch) ----
__device__ __forceinline__ void tdm_load_2d_f16(const void* gp, unsigned lds,
    unsigned dim0, unsigned dim1, unsigned til0, unsigned til1,
    unsigned stride0, unsigned padIv, unsigned padAmt) {
  const unsigned long long ga = (unsigned long long)(size_t)gp;
  v4u g0;
  g0[0] = 1u;                                                   // count=1
  g0[1] = lds;                                                  // lds_addr
  g0[2] = (unsigned)ga;                                         // addr[31:0]
  g0[3] = (unsigned)((ga >> 32) & 0x01FFFFFFull) | (2u << 30);  // addr[56:32]|type=2
  v8i_ g1;
  // data_size=1 (2B) | pad_enable | pad_interval | pad_amount
  g1[0] = (int)((1u << 16) | (1u << 20) | (padIv << 22) | (padAmt << 25));
  g1[1] = (int)(dim0 << 16);
  g1[2] = (int)((dim0 >> 16) | (dim1 << 16));
  g1[3] = (int)((dim1 >> 16) | (til0 << 16));
  g1[4] = (int)(til1);                                          // tile_dim2 = 0
  g1[5] = (int)(stride0);
  g1[6] = 0; g1[7] = 0;
  v4i_ gz4 = {0, 0, 0, 0};
  v8i_ gz8 = {0, 0, 0, 0, 0, 0, 0, 0};
  __builtin_amdgcn_tensor_load_to_lds(g0, g1, gz4, gz4, gz8, 0);
}
#endif

// ---------------- pre-pass 1: K f32 -> f16 (same layout) ----------------
struct __align__(16) H8 { _Float16 v[8]; };
__global__ __launch_bounds__(256)
void cvt_f32_f16(const float* __restrict__ src, _Float16* __restrict__ dst) {
  const int i = (blockIdx.x * 256 + threadIdx.x) * 8;
  float4 a = *(const float4*)(src + i);
  float4 b = *(const float4*)(src + i + 4);
  H8 o;
  o.v[0] = (_Float16)a.x; o.v[1] = (_Float16)a.y;
  o.v[2] = (_Float16)a.z; o.v[3] = (_Float16)a.w;
  o.v[4] = (_Float16)b.x; o.v[5] = (_Float16)b.y;
  o.v[6] = (_Float16)b.z; o.v[7] = (_Float16)b.w;
  *(H8*)(dst + i) = o;
}

// ---------------- pre-pass 2: V f32 -> f16, transposed per (b,h) ---------
// VT16[bh][f][s] = V[bh][s][f]
__global__ __launch_bounds__(128)
void cvt_vt(const float* __restrict__ V, _Float16* __restrict__ VT) {
  __shared__ __align__(16) _Float16 sT[D_][40];
  const int bh = blockIdx.y;
  const int kb = blockIdx.x * KB;
  const int tid = threadIdx.x;
  const float* Vb = V + (size_t)bh * S_ * D_;
  const int f0 = (tid & 15) * 4;
  const int k0 = (tid >> 4) * 4;
  float4 rr[4];
#pragma unroll
  for (int i = 0; i < 4; ++i)
    rr[i] = *(const float4*)(Vb + (size_t)(kb + k0 + i) * D_ + f0);
#pragma unroll
  for (int ff = 0; ff < 4; ++ff) {
    _Float16 tmp[4];
    tmp[0] = (_Float16)((&rr[0].x)[ff]);
    tmp[1] = (_Float16)((&rr[1].x)[ff]);
    tmp[2] = (_Float16)((&rr[2].x)[ff]);
    tmp[3] = (_Float16)((&rr[3].x)[ff]);
    *(uint2*)&sT[f0 + ff][k0] = *(const uint2*)tmp;
  }
  __syncthreads();
  const int row = tid >> 1;            // feature row
  const int ch  = (tid & 1) * 16;      // 16-key chunk
  _Float16* dst = VT + (size_t)bh * S_ * D_ + (size_t)row * S_ + kb + ch;
  const _Float16* sp = &sT[row][ch];
  *(uint4*)dst       = *(const uint4*)sp;
  *(uint4*)(dst + 8) = *(const uint4*)(sp + 8);
}

#if HAVE_TDM
// ================= fast kernel: all-f16 tiles via TDM ====================
__global__ __launch_bounds__(128)
void moe_attn_tdm(const float* __restrict__ Q, const _Float16* __restrict__ K16,
                  const _Float16* __restrict__ VT16, const float* __restrict__ route,
                  const float* __restrict__ mask, float* __restrict__ out) {
  __shared__ __align__(16) _Float16 sK[2][KB][72];   // 144B pitch (TDM pads 8 halves/64)
  __shared__ __align__(16) _Float16 sVT[2][D_][40];  // 80B pitch (TDM pads 8 halves/32)
  __shared__ __align__(32) _Float16 sP[WAVES][32][16];

  const int tid  = threadIdx.x;
  const int w    = tid >> 5;
  const int lane = tid & 31;
  const int n0   = lane & 15;
  const int h    = lane >> 4;

  const int bh    = blockIdx.y;
  const int b     = bh / H_;
  const int qTile = blockIdx.x * (WAVES * QT) + w * QT;

  const float*    Qb = Q    + (size_t)bh * S_ * D_;
  const _Float16* Kb = K16  + (size_t)bh * S_ * D_;
  const _Float16* Vt = VT16 + (size_t)bh * S_ * D_;
  const float*    rb = route + (size_t)b * E_ * S_;
  const float*    mb = mask  + (size_t)b * S_;
  float*          ob = out   + (size_t)bh * S_ * D_;

  auto issueTiles = [&](int kb, int buf) {
    if (w == 0)   // K tile: 32 keys x 64 feats, row = 64 halves -> pad to 72
      tdm_load_2d_f16(Kb + (size_t)kb * D_, (unsigned)(size_t)&sK[buf][0][0],
                      64u, 32u, 64u, 32u, 64u, /*padIv=*/4u, /*padAmt=*/3u);
    if (w == 1)   // V^T tile: 64 feats x 32 keys, row = 32 halves -> pad to 40
      tdm_load_2d_f16(Vt + kb, (unsigned)(size_t)&sVT[buf][0][0],
                      32u, 64u, 32u, 64u, (unsigned)S_, /*padIv=*/3u, /*padAmt=*/3u);
  };

  // ---- Q A-fragments, pre-scaled by gate * 1/sqrt(D) ----
  v16h aQ0, aQ1;
  {
    const int    qrow = qTile + n0;
    const float* qp   = Qb + (size_t)qrow * D_;
#pragma unroll
    for (int c = 0; c < 2; ++c) {
      const float g_lo = rb[(h + 4 * c) * S_ + qrow] * 0.125f;
      const float g_hi = rb[(2 + h + 4 * c) * S_ + qrow] * 0.125f;
#pragma unroll
      for (int j = 0; j < 16; ++j) {
        const int   f = (j & 7) + 16 * (j >> 3) + 8 * h + 32 * c;
        const float g = (j < 8) ? g_lo : g_hi;
        if (c == 0) aQ0[j] = (_Float16)(qp[f] * g);
        else        aQ1[j] = (_Float16)(qp[f] * g);
      }
    }
  }

  v16h bOne;
#pragma unroll
  for (int j = 0; j < 16; ++j) bOne[j] = (_Float16)1.0f;

  v8f Lacc = {};
  v8f O[4] = {};

  issueTiles(0, 0);
  __builtin_amdgcn_s_wait_tensorcnt(0);
  __syncthreads();

  for (int it = 0; it < NTILE; ++it) {
    const int  kb   = it * KB;
    const int  cur  = it & 1;
    const int  nxt  = cur ^ 1;
    const bool pref = (it + 1) < NTILE;

    if (pref) issueTiles(kb + KB, nxt);   // DMA of next tile overlaps compute

    const float penL0 = -1.442695e6f * (1.0f - mb[kb + n0]);
    const float penL1 = -1.442695e6f * (1.0f - mb[kb + 16 + n0]);

    // ---- scores ----
    v8f S0, S1;
    {
      v16h bK;
      v8f  z = {};
      const _Float16* kp = &sK[cur][n0][8 * h];
#pragma unroll
      for (int j = 0; j < 16; ++j) bK[j] = kp[(j & 7) + 16 * (j >> 3)];
      S0 = __builtin_amdgcn_wmma_f32_16x16x32_f16(false, aQ0, false, bK, (short)0, z, false, false);
      kp = &sK[cur][n0][32 + 8 * h];
#pragma unroll
      for (int j = 0; j < 16; ++j) bK[j] = kp[(j & 7) + 16 * (j >> 3)];
      S0 = __builtin_amdgcn_wmma_f32_16x16x32_f16(false, aQ1, false, bK, (short)0, S0, false, false);

      kp = &sK[cur][16 + n0][8 * h];
#pragma unroll
      for (int j = 0; j < 16; ++j) bK[j] = kp[(j & 7) + 16 * (j >> 3)];
      S1 = __builtin_amdgcn_wmma_f32_16x16x32_f16(false, aQ0, false, bK, (short)0, z, false, false);
      kp = &sK[cur][16 + n0][32 + 8 * h];
#pragma unroll
      for (int j = 0; j < 16; ++j) bK[j] = kp[(j & 7) + 16 * (j >> 3)];
      S1 = __builtin_amdgcn_wmma_f32_16x16x32_f16(false, aQ1, false, bK, (short)0, S1, false, false);
    }

    // ---- p = exp(score): scores bounded ~|4|, masked cols -> exp(-1e6)=0 ----
    _Float16* sPw = &sP[w][0][0];
#pragma unroll
    for (int r = 0; r < 8; ++r) {
      const float p0 = __builtin_amdgcn_exp2f(S0[r] * 1.44269504f + penL0);
      const float p1 = __builtin_amdgcn_exp2f(S1[r] * 1.44269504f + penL1);
      const int lr = (r + 8 * h) + 16 * (n0 >> 3);
      sPw[lr * 16 + (n0 & 7)]     = (_Float16)p0;
      sPw[lr * 16 + (n0 & 7) + 8] = (_Float16)p1;
    }
    asm volatile("s_wait_dscnt 0x0" ::: "memory");
    const v16h aP = *(const v16h*)&sP[w][lane][0];

    // ---- denominator + O += P*V : five WMMAs sharing A ----
    Lacc = __builtin_amdgcn_wmma_f32_16x16x32_f16(false, aP, false, bOne, (short)0, Lacc, false, false);
#pragma unroll
    for (int nt = 0; nt < 4; ++nt) {
      const _Float16* vp = &sVT[cur][16 * nt + n0][8 * h];
      v16h bV;
#pragma unroll
      for (int j = 0; j < 16; ++j) bV[j] = vp[(j & 7) + 16 * (j >> 3)];
      O[nt] = __builtin_amdgcn_wmma_f32_16x16x32_f16(false, aP, false, bV, (short)0, O[nt], false, false);
    }

    if (pref) __builtin_amdgcn_s_wait_tensorcnt(0);
    __syncthreads();
  }

#pragma unroll
  for (int r = 0; r < 8; ++r) {
    const int   qrow = qTile + r + 8 * h;
    const float inv  = __builtin_amdgcn_rcpf(Lacc[r]);
    float*      op   = ob + (size_t)qrow * D_;
#pragma unroll
    for (int nt = 0; nt < 4; ++nt) {
      const int   f = 16 * nt + n0;
      const float g = rb[(f >> 3) * S_ + qrow];
      op[f] = O[nt][r] * inv * g;
    }
  }
}
#endif  // HAVE_TDM

// ================= fallback kernel (self-contained, round-3 version) ======
__global__ __launch_bounds__(128)
void moe_flash_attn_fb(const float* __restrict__ Q, const float* __restrict__ K,
                       const float* __restrict__ V, const float* __restrict__ route,
                       const float* __restrict__ mask, float* __restrict__ out) {
  __shared__ __align__(16) float    sK[2][KB][68];
  __shared__ __align__(16) _Float16 sVT[2][D_][40];
  __shared__ __align__(32) _Float16 sP[WAVES][32][16];

  const int tid  = threadIdx.x;
  const int w    = tid >> 5;
  const int lane = tid & 31;
  const int n0   = lane & 15;
  const int h    = lane >> 4;

  const int bh    = blockIdx.y;
  const int b     = bh / H_;
  const int qTile = blockIdx.x * (WAVES * QT) + w * QT;

  const float* Qb = Q + (size_t)bh * S_ * D_;
  const float* Kb = K + (size_t)bh * S_ * D_;
  const float* Vb = V + (size_t)bh * S_ * D_;
  const float* rb = route + (size_t)b * E_ * S_;
  const float* mb = mask  + (size_t)b * S_;
  float*       ob = out   + (size_t)bh * S_ * D_;

  auto issueK = [&](int kb, int buf) {
#pragma unroll
    for (int i = 0; i < 4; ++i) {
      const int idx = tid + 128 * i;
      const int row = idx >> 4;
      const int cc  = idx & 15;
      unsigned     lds = (unsigned)(size_t)&sK[buf][row][cc * 4];
      const float* gp  = Kb + (size_t)(kb + row) * D_ + cc * 4;
      asm volatile("global_load_async_to_lds_b128 %0, %1, off"
                   :: "v"(lds), "v"(gp) : "memory");
    }
  };
  auto waitK = [&]() { asm volatile("s_wait_asynccnt 0x0" ::: "memory"); };

  const int vF0 = (tid & 15) * 4;
  const int vK0 = (tid >> 4) * 4;
  auto loadV = [&](int kb, float4* rr) {
#pragma unroll
    for (int i = 0; i < 4; ++i)
      rr[i] = *(const float4*)(Vb + (size_t)(kb + vK0 + i) * D_ + vF0);
  };
  auto storeV = [&](int buf, const float4* rr) {
#pragma unroll
    for (int ff = 0; ff < 4; ++ff) {
      _Float16 tmp[4];
      tmp[0] = (_Float16)((&rr[0].x)[ff]);
      tmp[1] = (_Float16)((&rr[1].x)[ff]);
      tmp[2] = (_Float16)((&rr[2].x)[ff]);
      tmp[3] = (_Float16)((&rr[3].x)[ff]);
      *(uint2*)&sVT[buf][vF0 + ff][vK0] = *(const uint2*)tmp;
    }
  };

  v16h aQ0, aQ1;
  {
    const int    qrow = qTile + n0;
    const float* qp   = Qb + (size_t)qrow * D_;
#pragma unroll
    for (int c = 0; c < 2; ++c) {
      const float g_lo = rb[(h + 4 * c) * S_ + qrow] * 0.125f;
      const float g_hi = rb[(2 + h + 4 * c) * S_ + qrow] * 0.125f;
#pragma unroll
      for (int j = 0; j < 16; ++j) {
        const int   f = (j & 7) + 16 * (j >> 3) + 8 * h + 32 * c;
        const float g = (j < 8) ? g_lo : g_hi;
        if (c == 0) aQ0[j] = (_Float16)(qp[f] * g);
        else        aQ1[j] = (_Float16)(qp[f] * g);
      }
    }
  }

  v16h bOne;
#pragma unroll
  for (int j = 0; j < 16; ++j) bOne[j] = (_Float16)1.0f;

  v8f Lacc = {};
  v8f O[4] = {};

  {
    float4 r0v[4];
    issueK(0, 0);
    loadV(0, r0v);
    storeV(0, r0v);
    waitK();
  }
  __syncthreads();

  for (int it = 0; it < NTILE; ++it) {
    const int  kb   = it * KB;
    const int  cur  = it & 1;
    const int  nxt  = cur ^ 1;
    const bool pref = (it + 1) < NTILE;

    float4 rr[4];
    if (pref) {
      issueK(kb + KB, nxt);
      loadV(kb + KB, rr);
    }
    const float penL0 = -1.442695e6f * (1.0f - mb[kb + n0]);
    const float penL1 = -1.442695e6f * (1.0f - mb[kb + 16 + n0]);

    v8f S0, S1;
    {
      v16h bK;
      v8f  z = {};
      const float* kp = &sK[cur][n0][8 * h];
#pragma unroll
      for (int j = 0; j < 16; ++j) bK[j] = (_Float16)kp[(j & 7) + 16 * (j >> 3)];
      S0 = __builtin_amdgcn_wmma_f32_16x16x32_f16(false, aQ0, false, bK, (short)0, z, false, false);
      kp = &sK[cur][n0][32 + 8 * h];
#pragma unroll
      for (int j = 0; j < 16; ++j) bK[j] = (_Float16)kp[(j & 7) + 16 * (j >> 3)];
      S0 = __builtin_amdgcn_wmma_f32_16x16x32_f16(false, aQ1, false, bK, (short)0, S0, false, false);

      kp = &sK[cur][16 + n0][8 * h];
#pragma unroll
      for (int j = 0; j < 16; ++j) bK[j] = (_Float16)kp[(j & 7) + 16 * (j >> 3)];
      S1 = __builtin_amdgcn_wmma_f32_16x16x32_f16(false, aQ0, false, bK, (short)0, z, false, false);
      kp = &sK[cur][16 + n0][32 + 8 * h];
#pragma unroll
      for (int j = 0; j < 16; ++j) bK[j] = (_Float16)kp[(j & 7) + 16 * (j >> 3)];
      S1 = __builtin_amdgcn_wmma_f32_16x16x32_f16(false, aQ1, false, bK, (short)0, S1, false, false);
    }

    _Float16* sPw = &sP[w][0][0];
#pragma unroll
    for (int r = 0; r < 8; ++r) {
      const float p0 = __builtin_amdgcn_exp2f(S0[r] * 1.44269504f + penL0);
      const float p1 = __builtin_amdgcn_exp2f(S1[r] * 1.44269504f + penL1);
      const int lr = (r + 8 * h) + 16 * (n0 >> 3);
      sPw[lr * 16 + (n0 & 7)]     = (_Float16)p0;
      sPw[lr * 16 + (n0 & 7) + 8] = (_Float16)p1;
    }
    asm volatile("s_wait_dscnt 0x0" ::: "memory");
    const v16h aP = *(const v16h*)&sP[w][lane][0];

    Lacc = __builtin_amdgcn_wmma_f32_16x16x32_f16(false, aP, false, bOne, (short)0, Lacc, false, false);
#pragma unroll
    for (int nt = 0; nt < 4; ++nt) {
      const _Float16* vp = &sVT[cur][16 * nt + n0][8 * h];
      v16h bV;
#pragma unroll
      for (int j = 0; j < 16; ++j) bV[j] = vp[(j & 7) + 16 * (j >> 3)];
      O[nt] = __builtin_amdgcn_wmma_f32_16x16x32_f16(false, aP, false, bV, (short)0, O[nt], false, false);
    }

    if (pref) {
      storeV(nxt, rr);
      waitK();
    }
    __syncthreads();
  }

#pragma unroll
  for (int r = 0; r < 8; ++r) {
    const int   qrow = qTile + r + 8 * h;
    const float inv  = __builtin_amdgcn_rcpf(Lacc[r]);
    float*      op   = ob + (size_t)qrow * D_;
#pragma unroll
    for (int nt = 0; nt < 4; ++nt) {
      const int   f = 16 * nt + n0;
      const float g = rb[(f >> 3) * S_ + qrow];
      op[f] = O[nt][r] * inv * g;
    }
  }
}

extern "C" void kernel_launch(void* const* d_in, const int* in_sizes, int n_in,
                              void* d_out, int out_size, void* d_ws, size_t ws_size,
                              hipStream_t stream) {
  (void)in_sizes; (void)n_in; (void)out_size;
  const float* Q  = (const float*)d_in[0];
  const float* K  = (const float*)d_in[1];
  const float* V  = (const float*)d_in[2];
  const float* rm = (const float*)d_in[3];
  const float* mk = (const float*)d_in[4];
  float*       o  = (float*)d_out;

#if HAVE_TDM
  const size_t need = 2ull * B_ * H_ * S_ * D_ * 2ull;  // K16 + VT16 (f16)
  if (ws_size >= need) {
    _Float16* k16  = (_Float16*)d_ws;
    _Float16* vt16 = k16 + (size_t)B_ * H_ * S_ * D_;
    const int n = B_ * H_ * S_ * D_;
    cvt_f32_f16<<<n / 2048, 256, 0, stream>>>(K, k16);
    cvt_vt<<<dim3(S_ / KB, B_ * H_), 128, 0, stream>>>(V, vt16);
    dim3 grid(S_ / (WAVES * QT), B_ * H_);
    moe_attn_tdm<<<grid, WAVES * 32, 0, stream>>>(Q, k16, vt16, rm, mk, o);
    return;
  }
#endif
  (void)d_ws; (void)ws_size;
  dim3 grid(S_ / (WAVES * QT), B_ * H_);
  moe_flash_attn_fb<<<grid, WAVES * 32, 0, stream>>>(Q, K, V, rm, mk, o);
}
